// MultiheadAttention_24970939859004
// MI455X (gfx1250) — matrix operand order repaired and checked by reading, not compiled, not run on hardware
//
#include <hip/hip_runtime.h>

// ---------------------------------------------------------------------------
// MultiheadAttention forward for gfx1250 (MI455X).
//   B=4, S=2048, D=1024, H=16, DK=64
// GEMMs use v_wmma_f32_16x16x32_bf16 (fp32 inputs converted to bf16, f32 acc).
// Outputs: attn (B,S,D) f32 then sim_mean (B,S,S) f32, concatenated in d_out.
//
// Round-2 tiling: 16x64 register tiles in the dense GEMMs (4 WMMAs per
// A-fragment load), Q-fragment reuse across k-tiles in sim_kernel, and
// 32-key sweep-1 iterations in the attention core.
// ---------------------------------------------------------------------------

#define BB 4
#define SS 2048
#define DD 1024
#define HH 16
#define DKK 64

typedef __attribute__((ext_vector_type(16))) __bf16 v16bf;
typedef __attribute__((ext_vector_type(8)))  float  v8f;

union V16BF { uint4 u[2]; v16bf v; };

__device__ __forceinline__ v8f wmma_bf16(v16bf a, v16bf b, v8f c) {
  return __builtin_amdgcn_wmma_f32_16x16x32_bf16(
      /*neg_a=*/false, a, /*neg_b=*/false, b,
      /*c_mod=*/(short)0, c, /*reuse_a=*/false, /*reuse_b=*/false);
}

// A-fragment (16x32, row-major source, bf16). ISA layout:
// lanes 0-15: M=lane, K = {0..7, 16..23}; lanes 16-31: M=lane-16, K = {8..15, 24..31}
__device__ __forceinline__ v16bf load_a_frag_bf16(const __bf16* base, int ld, int lane) {
  const int m = lane & 15;
  const int c = (lane >> 4) << 3;          // 0 or 8
  const __bf16* p = base + (size_t)m * ld + c;
  V16BF r;
  r.u[0] = *(const uint4*)(p);             // K = c .. c+7
  r.u[1] = *(const uint4*)(p + 16);        // K = c+16 .. c+23
  return r.v;
}

__device__ __forceinline__ v16bf load_a_frag_f32(const float* base, int ld, int lane) {
  const int m = lane & 15;
  const int c = (lane >> 4) << 3;
  const float* p = base + (size_t)m * ld + c;
  v16bf a;
#pragma unroll
  for (int i = 0; i < 8; ++i) a[i] = (__bf16)p[i];
#pragma unroll
  for (int i = 0; i < 8; ++i) a[8 + i] = (__bf16)p[16 + i];
  return a;
}

// B-fragment (32x16). Column n = lane&15 lives at base + n*ld; contraction
// range: lanes 0-15 -> K {0..7, 16..23}, lanes 16-31 -> K {8..15, 24..31}.
__device__ __forceinline__ v16bf load_b_frag_bf16(const __bf16* base, int ld, int lane) {
  const int n  = lane & 15;
  const int kb = (lane >> 4) << 3;         // 0 or 8
  const __bf16* p = base + (size_t)n * ld + kb;
  V16BF r;
  r.u[0] = *(const uint4*)(p);
  r.u[1] = *(const uint4*)(p + 16);
  return r.v;
}

__device__ __forceinline__ v16bf load_b_frag_f32(const float* base, int ld, int lane) {
  const int n  = lane & 15;
  const int kb = (lane >> 4) << 3;
  const float* p = base + (size_t)n * ld + kb;
  v16bf b;
#pragma unroll
  for (int i = 0; i < 8; ++i) b[i] = (__bf16)p[i];
#pragma unroll
  for (int i = 0; i < 8; ++i) b[8 + i] = (__bf16)p[16 + i];
  return b;
}

// ---------------------------------------------------------------------------
// Kernel 1: QKV projections. X (8192 x 1024) @ W^T -> bf16.
// One wave computes a 16x64 output tile (one full head's columns): the A
// fragment is reused across 4 WMMAs per 32-deep contraction step.
// Q,K layout (B,H,S,DK); V stored transposed (B,H,DK,S).
// ---------------------------------------------------------------------------
__global__ __launch_bounds__(32) void proj_kernel(
    const float* __restrict__ q, const float* __restrict__ k,
    const float* __restrict__ v, const float* __restrict__ w,
    __bf16* __restrict__ Qb, __bf16* __restrict__ Kb, __bf16* __restrict__ Vt)
{
  const int lane = threadIdx.x;
  const int NT4 = DD / 64;                 // 16 column-groups of 64
  const int MT  = (BB * SS) / 16;          // 512
  int tile  = blockIdx.x;                  // 3 * MT * NT4
  int which = tile / (MT * NT4);
  int rem   = tile % (MT * NT4);
  int mt = rem / NT4, nt4 = rem % NT4;     // nt4 == head index

  const float* X = (which == 0) ? q : ((which == 1) ? k : v);
  const float* W = w + (size_t)which * DD * DD;
  const float* arow = X + (size_t)(mt * 16) * DD;
  const float* brow = W + (size_t)(nt4 * 64) * DD;

  v8f acc0 = {}, acc1 = {}, acc2 = {}, acc3 = {};
#pragma unroll 2
  for (int kk = 0; kk < DD; kk += 32) {
    v16bf a = load_a_frag_f32(arow + kk, DD, lane);
    acc0 = wmma_bf16(a, load_b_frag_f32(brow + (size_t) 0 * DD + kk, DD, lane), acc0);
    acc1 = wmma_bf16(a, load_b_frag_f32(brow + (size_t)16 * DD + kk, DD, lane), acc1);
    acc2 = wmma_bf16(a, load_b_frag_f32(brow + (size_t)32 * DD + kk, DD, lane), acc2);
    acc3 = wmma_bf16(a, load_b_frag_f32(brow + (size_t)48 * DD + kk, DD, lane), acc3);
  }

  const int h    = nt4;                    // whole 64-wide tile = one head
  const int moff = (lane >> 4) * 8;
#pragma unroll
  for (int j = 0; j < 4; ++j) {
    v8f a = (j == 0) ? acc0 : (j == 1) ? acc1 : (j == 2) ? acc2 : acc3;
    int dcol = j * 16 + (lane & 15);       // dim inside head
#pragma unroll
    for (int r = 0; r < 8; ++r) {
      int m = mt * 16 + moff + r;          // token id
      int b = m >> 11;                     // m / S
      int s = m & (SS - 1);
      __bf16 val = (__bf16)a[r];
      if (which == 2) {
        Vt[((size_t)(b * HH + h) * DKK + dcol) * SS + s] = val;
      } else {
        __bf16* dst = (which == 0) ? Qb : Kb;
        dst[((size_t)(b * HH + h) * SS + s) * DKK + dcol] = val;
      }
    }
  }
}

// ---------------------------------------------------------------------------
// Kernel 2: attention core. One wave per (b,h,q-tile of 16 rows).
// Sweep 1: scores (32 keys/iter) -> row max/sumexp (saved to ws).
// Sweep 2: P -> O = P @ V (4 WMMAs per 32-key P fragment).
// ---------------------------------------------------------------------------
__global__ __launch_bounds__(32) void attn_kernel(
    const __bf16* __restrict__ Qb, const __bf16* __restrict__ Kb,
    const __bf16* __restrict__ Vt,
    float* __restrict__ Mrow, float* __restrict__ Lrow,
    __bf16* __restrict__ Ob)
{
  __shared__ float lds_s[16 * 34];         // 16 x 32 score block (padded)
  __shared__ float lds_p[16 * 36];         // P tile, 16 x 32 (padded stride)
  __shared__ float sm[16], sinvl[16];

  const int lane = threadIdx.x;
  const int QT = SS / 16;                  // 128 k-tiles
  int blk = blockIdx.x;                    // B*H*QT
  int qt  = blk % QT;
  int bh  = blk / QT;                      // b*H + h
  const int b = bh >> 4, h = bh & 15;

  const __bf16* Qbase = Qb + ((size_t)bh * SS + qt * 16) * DKK;
  const __bf16* Kbase = Kb + (size_t)bh * SS * DKK;
  const __bf16* Vbase = Vt + (size_t)bh * DKK * SS;
  const float scale = 0.125f;              // 1/sqrt(DK)

  // Q A-fragments for the two 32-wide contraction chunks of DK=64.
  v16bf qa0 = load_a_frag_bf16(Qbase, DKK, lane);
  v16bf qa1 = load_a_frag_bf16(Qbase + 32, DKK, lane);

  const int n    = lane & 15;
  const int moff = (lane >> 4) * 8;
  float m_run = -3.0e38f, l_run = 0.0f;    // live in lanes 0..15

  // ---- Sweep 1: row max / sumexp, 32 keys per iteration ------------------
  for (int kt2 = 0; kt2 < QT / 2; ++kt2) {
    const __bf16* Kt0 = Kbase + (size_t)(kt2 * 32) * DKK;
    if (kt2 + 1 < QT / 2)
      __builtin_prefetch((const void*)(Kt0 + 32 * DKK), 0, 1);
#pragma unroll
    for (int sub = 0; sub < 2; ++sub) {
      const __bf16* Kt = Kt0 + (size_t)(sub * 16) * DKK;
      v8f c = {};
      c = wmma_bf16(qa0, load_b_frag_bf16(Kt,      DKK, lane), c);
      c = wmma_bf16(qa1, load_b_frag_bf16(Kt + 32, DKK, lane), c);
#pragma unroll
      for (int r = 0; r < 8; ++r)
        lds_s[(moff + r) * 34 + sub * 16 + n] = c[r];
    }
    __syncthreads();
    if (lane < 16) {
      const float* rowp = &lds_s[lane * 34];
      float tmax = -3.0e38f;
#pragma unroll
      for (int j = 0; j < 32; ++j) tmax = fmaxf(tmax, rowp[j]);
      tmax *= scale;
      float mnew = fmaxf(m_run, tmax);
      float ssum = 0.0f;
#pragma unroll
      for (int j = 0; j < 32; ++j) ssum += __expf(rowp[j] * scale - mnew);
      l_run = l_run * __expf(m_run - mnew) + ssum;
      m_run = mnew;
    }
    __syncthreads();
  }
  if (lane < 16) {
    sm[lane]    = m_run;
    sinvl[lane] = 1.0f / l_run;
    Mrow[(size_t)bh * SS + qt * 16 + lane] = m_run;
    Lrow[(size_t)bh * SS + qt * 16 + lane] = l_run;
  }
  __syncthreads();

  // ---- Sweep 2: O = P @ V -------------------------------------------------
  v8f o0 = {}, o1 = {}, o2 = {}, o3 = {};
  for (int kt2 = 0; kt2 < QT / 2; ++kt2) {
#pragma unroll
    for (int sub = 0; sub < 2; ++sub) {
      int kt = kt2 * 2 + sub;
      const __bf16* Kt = Kbase + (size_t)(kt * 16) * DKK;
      v8f c = {};
      c = wmma_bf16(qa0, load_b_frag_bf16(Kt,      DKK, lane), c);
      c = wmma_bf16(qa1, load_b_frag_bf16(Kt + 32, DKK, lane), c);
#pragma unroll
      for (int r = 0; r < 8; ++r) {
        int row = moff + r;
        float p = __expf(c[r] * scale - sm[row]) * sinvl[row];
        lds_p[row * 36 + sub * 16 + n] = p;
      }
    }
    __syncthreads();
    // P as bf16 A-fragment (16x32) out of LDS (cross-lane transpose).
    v16bf pa;
    {
      const int mm = lane & 15;
      const int cc = (lane >> 4) << 3;
      const float* pp = &lds_p[mm * 36 + cc];
#pragma unroll
      for (int i = 0; i < 8; ++i) pa[i]     = (__bf16)pp[i];
#pragma unroll
      for (int i = 0; i < 8; ++i) pa[8 + i] = (__bf16)pp[16 + i];
    }
    const __bf16* Vk = Vbase + kt2 * 32;   // keys kt2*32 .. +31, d-major rows
    o0 = wmma_bf16(pa, load_b_frag_bf16(Vk + (size_t)0  * SS, SS, lane), o0);
    o1 = wmma_bf16(pa, load_b_frag_bf16(Vk + (size_t)16 * SS, SS, lane), o1);
    o2 = wmma_bf16(pa, load_b_frag_bf16(Vk + (size_t)32 * SS, SS, lane), o2);
    o3 = wmma_bf16(pa, load_b_frag_bf16(Vk + (size_t)48 * SS, SS, lane), o3);
    __syncthreads();
  }

  // Store O (bf16) into (B,S,D) layout for the output GEMM.
#pragma unroll
  for (int t = 0; t < 4; ++t) {
    v8f o = (t == 0) ? o0 : (t == 1) ? o1 : (t == 2) ? o2 : o3;
    int col = h * DKK + t * 16 + n;
#pragma unroll
    for (int r = 0; r < 8; ++r) {
      int s   = qt * 16 + moff + r;
      size_t tok = (size_t)b * SS + s;
      Ob[tok * DD + col] = (__bf16)o[r];
    }
  }
}

// ---------------------------------------------------------------------------
// Kernel 3: sim_mean = mean over heads of softmax(scores).
// h-outer / k-tile-inner: Q fragments and m/l loaded once per head,
// 8 per-tile accumulators held in registers. Deterministic (no atomics).
// ---------------------------------------------------------------------------
__global__ __launch_bounds__(32) void sim_kernel(
    const __bf16* __restrict__ Qb, const __bf16* __restrict__ Kb,
    const float* __restrict__ Mrow, const float* __restrict__ Lrow,
    float* __restrict__ sim)
{
  __shared__ float sm[16], sinvl[16];
  const int lane = threadIdx.x;
  const int QT = SS / 16;                  // 128
  const int KSPLIT = 16, KPER = QT / KSPLIT;   // 8 k-tiles per block
  int blk = blockIdx.x;                    // B * QT * KSPLIT
  int ks  = blk % KSPLIT;
  int rem = blk / KSPLIT;
  int qt  = rem % QT;
  int b   = rem / QT;

  const int n    = lane & 15;
  const int moff = (lane >> 4) * 8;
  const float scale = 0.125f;
  const float hinv  = 1.0f / (float)HH;

  v8f ps[KPER];
#pragma unroll
  for (int kk = 0; kk < KPER; ++kk) ps[kk] = (v8f){};

  for (int h = 0; h < HH; ++h) {
    int bh = b * HH + h;
    __syncthreads();                       // protect sm/sinvl readers
    if (lane < 16) {
      sm[lane]    = Mrow[(size_t)bh * SS + qt * 16 + lane];
      sinvl[lane] = 1.0f / Lrow[(size_t)bh * SS + qt * 16 + lane];
    }
    __syncthreads();
    const __bf16* Qbase = Qb + ((size_t)bh * SS + qt * 16) * DKK;
    v16bf qa0 = load_a_frag_bf16(Qbase,      DKK, lane);
    v16bf qa1 = load_a_frag_bf16(Qbase + 32, DKK, lane);
#pragma unroll 2
    for (int kk = 0; kk < KPER; ++kk) {
      int kt = ks * KPER + kk;
      const __bf16* Kt = Kb + ((size_t)bh * SS + kt * 16) * DKK;
      v8f c = {};
      c = wmma_bf16(qa0, load_b_frag_bf16(Kt,      DKK, lane), c);
      c = wmma_bf16(qa1, load_b_frag_bf16(Kt + 32, DKK, lane), c);
#pragma unroll
      for (int r = 0; r < 8; ++r) {
        int row = moff + r;
        ps[kk][r] += __expf(c[r] * scale - sm[row]) * sinvl[row] * hinv;
      }
    }
  }

#pragma unroll
  for (int kk = 0; kk < KPER; ++kk) {
    int kt = ks * KPER + kk;
#pragma unroll
    for (int r = 0; r < 8; ++r) {
      int qrow = qt * 16 + moff + r;
      sim[((size_t)b * SS + qrow) * SS + kt * 16 + n] = ps[kk][r];
    }
  }
}

// ---------------------------------------------------------------------------
// Kernel 4: output projection  attn = O @ Wout^T   (f32 result).
// 16x64 register tile per wave: A fragment reused across 4 WMMAs.
// ---------------------------------------------------------------------------
__global__ __launch_bounds__(32) void outproj_kernel(
    const __bf16* __restrict__ Ob, const float* __restrict__ Wout,
    float* __restrict__ out)
{
  const int lane = threadIdx.x;
  const int NT4 = DD / 64;                 // 16
  int tile = blockIdx.x;                   // 512 * 16
  int mt = tile / NT4, nt4 = tile % NT4;

  const __bf16* arow = Ob   + (size_t)(mt * 16) * DD;
  const float*  brow = Wout + (size_t)(nt4 * 64) * DD;

  v8f acc0 = {}, acc1 = {}, acc2 = {}, acc3 = {};
#pragma unroll 2
  for (int kk = 0; kk < DD; kk += 32) {
    v16bf a = load_a_frag_bf16(arow + kk, DD, lane);
    acc0 = wmma_bf16(a, load_b_frag_f32(brow + (size_t) 0 * DD + kk, DD, lane), acc0);
    acc1 = wmma_bf16(a, load_b_frag_f32(brow + (size_t)16 * DD + kk, DD, lane), acc1);
    acc2 = wmma_bf16(a, load_b_frag_f32(brow + (size_t)32 * DD + kk, DD, lane), acc2);
    acc3 = wmma_bf16(a, load_b_frag_f32(brow + (size_t)48 * DD + kk, DD, lane), acc3);
  }

  const int moff = (lane >> 4) * 8;
#pragma unroll
  for (int j = 0; j < 4; ++j) {
    v8f a = (j == 0) ? acc0 : (j == 1) ? acc1 : (j == 2) ? acc2 : acc3;
    int nn = nt4 * 64 + j * 16 + (lane & 15);
#pragma unroll
    for (int r = 0; r < 8; ++r) {
      int m = mt * 16 + moff + r;
      out[(size_t)m * DD + nn] = a[r];
    }
  }
}

// ---------------------------------------------------------------------------
extern "C" void kernel_launch(void* const* d_in, const int* in_sizes, int n_in,
                              void* d_out, int out_size, void* d_ws, size_t ws_size,
                              hipStream_t stream) {
  (void)in_sizes; (void)n_in; (void)out_size; (void)ws_size;

  const float* q    = (const float*)d_in[0];
  const float* k    = (const float*)d_in[1];
  const float* v    = (const float*)d_in[2];
  const float* w_in = (const float*)d_in[3];   // (3,D,D)
  const float* w_out= (const float*)d_in[4];   // (D,D)

  float* attn_out = (float*)d_out;                           // (B,S,D)
  float* sim_out  = attn_out + (size_t)BB * SS * DD;         // (B,S,S)

  // Workspace layout (bf16 QKV+O, f32 m/l): ~65 MB.
  const size_t eQ = (size_t)BB * HH * SS * DKK;              // 8,388,608
  __bf16* Qb = (__bf16*)d_ws;
  __bf16* Kb = Qb + eQ;
  __bf16* Vt = Kb + eQ;
  __bf16* Ob = Vt + eQ;
  float*  Mrow = (float*)(Ob + eQ);
  float*  Lrow = Mrow + (size_t)BB * HH * SS;

  const int MT  = (BB * SS) / 16;  // 512
  const int NT4 = DD / 64;         // 16
  const int QT  = SS / 16;         // 128

  proj_kernel<<<3 * MT * NT4, 32, 0, stream>>>(q, k, v, w_in, Qb, Kb, Vt);
  attn_kernel<<<BB * HH * QT, 32, 0, stream>>>(Qb, Kb, Vt, Mrow, Lrow, Ob);
  sim_kernel<<<BB * QT * 16, 32, 0, stream>>>(Qb, Kb, Mrow, Lrow, sim_out);
  outproj_kernel<<<MT * NT4, 32, 0, stream>>>(Ob, w_out, attn_out);
}
